// GR_Weight_76519137346176
// MI455X (gfx1250) — matrix operand order repaired and checked by reading, not compiled, not run on hardware
//
#include <hip/hip_runtime.h>
#include <hip/hip_bf16.h>

// ---------------------------------------------------------------------------
// GR_A recurrent scan + Globel_state for MI455X (gfx1250, wave32, WMMA bf16)
//
//   kernel 1: init bf16 h double-buffer + grid barrier state
//   kernel 2: x f32 -> bf16 (one pass)
//   kernel 3: pack [Wx;Wh] (K=2048,N=4096) -> fragment-major bf16 tiles
//   kernel 4: pack Wgh     (K=1024,N=3072) -> fragment-major bf16 tiles
//   kernel 5/6: gz = g0@Wg + b ; gzg = g0@Wgg + bg
//   kernel 7: persistent cooperative scan, 64 WGs x 256 thr (8 waves):
//       double-buffered async global->LDS A staging (ASYNCcnt pipelined),
//       z = [x_t | h] @ [Wx;Wh] via v_wmma_f32_16x16x32_bf16, registerized
//       gate recurrence, one grid barrier/step; then h_fin@Wgh global state.
// ---------------------------------------------------------------------------

typedef __bf16 bf16x16 __attribute__((ext_vector_type(16)));
typedef float f32x8    __attribute__((ext_vector_type(8)));

#define T_STEPS 1024
#define BB      32
#define HH      1024
#define KTOT    2048      // IN + H fused K
#define KCHUNK  256
#define NCHUNK  (KTOT / KCHUNK)   // 8 chunks per step
#define APAD    8         // lds_a row pad (elems): 528B row -> conflict-free
#define NWG     64

__device__ __forceinline__ unsigned short f2bf(float f) {
  unsigned int u = __float_as_uint(f);
  unsigned int r = (u + 0x7FFFu + ((u >> 16) & 1u)) >> 16;  // RNE
  return (unsigned short)r;
}

__device__ __forceinline__ float sigf(float x) {
  return 1.0f / (1.0f + __expf(-x));
}

// -------------------------- init / convert ---------------------------------
__global__ void init_state_kernel(unsigned short* hbf0, unsigned short* hbf1,
                                  int* bar) {
  int i = blockIdx.x * blockDim.x + threadIdx.x;
  if (i < BB * HH) { hbf0[i] = 0; hbf1[i] = 0; }
  if (i < 2) bar[i] = 0;
}

__global__ void f32_to_bf16_kernel(const float* __restrict__ src,
                                   unsigned short* __restrict__ dst, int n) {
  int i = blockIdx.x * blockDim.x + threadIdx.x;
  if (i < n) dst[i] = f2bf(src[i]);
}

// ------------------- pack weights to B-fragment tiles ----------------------
// tile = kti*(N/16)+nti ; within tile: [lane 0..31][i 0..15]
// lane L, half i -> K = kti*32 + (L>>4)*16 + i ; n = nti*16 + (L&15)
__global__ void pack_b_tiles_kernel(const float* __restrict__ W0,
                                    const float* __restrict__ W1,
                                    int Krows0, int N,
                                    unsigned short* __restrict__ dst) {
  int tile = blockIdx.x;
  int nNt  = N >> 4;
  int kti  = tile / nNt;
  int nti  = tile % nNt;
  for (int e = threadIdx.x; e < 512; e += blockDim.x) {
    int lane = e >> 4;
    int i    = e & 15;
    int k    = kti * 32 + ((lane >> 4) << 4) + i;
    int n    = nti * 16 + (lane & 15);
    float v  = (k < Krows0) ? W0[(size_t)k * N + n]
                            : W1[(size_t)(k - Krows0) * N + n];
    dst[(size_t)tile * 512 + e] = f2bf(v);
  }
}

// ------------------------- gz = g @ W + bias -------------------------------
__global__ void colsum_kernel(const float* __restrict__ W,
                              const float* __restrict__ bias,
                              const float* __restrict__ g,
                              float* __restrict__ out, int K, int N) {
  int n = blockIdx.x * blockDim.x + threadIdx.x;
  if (n >= N) return;
  float s = bias[n];
  for (int k = 0; k < K; ++k) s += g[k] * W[(size_t)k * N + n];
  out[n] = s;
}

// ----------------------- grid-wide barrier --------------------------------
__device__ __forceinline__ void grid_barrier(int* bar) {
  __threadfence();
  __syncthreads();
  if (threadIdx.x == 0) {
    volatile int* vgen = bar + 1;
    int g = *vgen;
    int prev = atomicAdd(bar, 1);
    if (prev == NWG - 1) {
      bar[0] = 0;
      __threadfence();
      atomicAdd(bar + 1, 1);
    } else {
      while (*vgen == g) { __builtin_amdgcn_s_sleep(2); }
    }
  }
  __syncthreads();
}

// ------------------ async global -> LDS staging (ASYNCcnt) -----------------
__device__ __forceinline__ void async_copy_b128(unsigned lds_off,
                                                const unsigned short* gptr) {
  asm volatile("global_load_async_to_lds_b128 %0, %1, off"
               :: "v"(lds_off), "v"(gptr) : "memory");
}
template <int N>
__device__ __forceinline__ void wait_asynccnt() {
  asm volatile("s_wait_asynccnt %0" :: "i"(N) : "memory");
}

// Issue async copies for one [32 x KCHUNK] bf16 chunk (4 x b128 per thread).
__device__ __forceinline__ void stage_issue(unsigned buf_off,
                                            const unsigned short* __restrict__ src) {
  const int tid = threadIdx.x;
#pragma unroll
  for (int u = 0; u < 4; ++u) {
    int cid = tid + u * 256;          // 0..1023 16B chunks
    int row = cid >> 5;               // 32 chunks per row
    int o16 = cid & 31;
    async_copy_b128(buf_off + (unsigned)(row * (KCHUNK + APAD) + o16 * 8) * 2u,
                    src + (size_t)row * HH + o16 * 8);
  }
}

// ------------------ fragment helpers (contiguous k order, A and B alike) ------------------
// A 16x32 bf16: lane L -> M=L&15 ; vgpr v, half h -> K=v*2+h+(L>>4)*16
__device__ __forceinline__ bf16x16 load_a_frag(
    const unsigned short (*lds_a)[KCHUNK + APAD], int row, int klb, int lane) {
  union { bf16x16 v; unsigned int u[8]; } afr;
#pragma unroll
  for (int v8 = 0; v8 < 8; ++v8) {
    int kb = (v8 << 1) + ((lane >> 4) << 4);
    afr.u[v8] = *(const unsigned int*)&lds_a[row][klb + kb];
  }
  return afr.v;
}

// B 32x16 bf16: lane L, half i -> K=(L>>4)*16+i ; N=L&15 (matches pack layout)
__device__ __forceinline__ bf16x16 load_b_frag(
    const unsigned short* __restrict__ wtiles, size_t tileIdx, int lane) {
  union { bf16x16 v; uint4 q[2]; } bfr;
  const uint4* bp = (const uint4*)(wtiles + tileIdx * 512 + lane * 16);
  bfr.q[0] = bp[0];
  bfr.q[1] = bp[1];
  return bfr.v;
}

// ----------------------- persistent scan kernel ----------------------------
__global__ __launch_bounds__(256) void gr_scan_kernel(
    const unsigned short* __restrict__ xbf,  // [T,B,IN] bf16
    const float* __restrict__ a,             // [H]
    const unsigned short* __restrict__ wc,   // packed [64 kt][256 nt][512]
    const unsigned short* __restrict__ wgh,  // packed [32 kt][192 nt][512]
    const float* __restrict__ gz,            // [4H]
    const float* __restrict__ gzg,           // [3H]
    unsigned short* __restrict__ hbf0,       // [B,H] bf16 double buffer
    unsigned short* __restrict__ hbf1,
    const float* __restrict__ cg0,           // [H]
    float* __restrict__ out,                 // [2*B*H]: g_final then cg_new
    int* __restrict__ bar) {
  __shared__ __align__(16) unsigned short lds_a[2][BB][KCHUNK + APAD];
  __shared__ float lds_z[4][BB][17];

  const int wg   = blockIdx.x;        // owns h-cols [wg*16, wg*16+16)
  const int tid  = threadIdx.x;
  const int lane = tid & 31;
  const int wave = tid >> 5;
  unsigned aoff[2];
  aoff[0] = (unsigned)(unsigned long long)&lds_a[0][0][0];
  aoff[1] = (unsigned)(unsigned long long)&lds_a[1][0][0];

  const int gGate = wave >> 1;        // 0..3
  const int rHalf = wave & 1;         // 0..1
  const int arow  = rHalf * 16 + (lane & 15);

  // registerized per-thread recurrence state: 2 owned (b, j) elements
  int eb[2], ejj[2], ej[2];
  float gzi[2], gzf[2], gzo[2], gzc[2], av[2];
#pragma unroll
  for (int u = 0; u < 2; ++u) {
    int e  = tid + u * 256;
    eb[u]  = e >> 4;
    ejj[u] = e & 15;
    ej[u]  = wg * 16 + ejj[u];
    gzi[u] = gz[ej[u]];
    gzf[u] = gz[HH + ej[u]];
    gzo[u] = gz[2 * HH + ej[u]];
    gzc[u] = gz[3 * HH + ej[u]];
    av[u]  = a[ej[u]];
  }
  float hreg[2] = {0.0f, 0.0f};
  float creg[2] = {0.0f, 0.0f};

  // prologue: pre-issue chunk 0 (x) of step 0 into buffer 0
  stage_issue(aoff[0], xbf);

  for (int t = 0; t < T_STEPS; ++t) {
    const unsigned short* hcur = (t & 1) ? hbf1 : hbf0;
    unsigned short*       hnxt = (t & 1) ? hbf0 : hbf1;
    const unsigned short* xt   = xbf + (size_t)t * BB * HH;

    f32x8 acc = {};
    const int ntile = gGate * (HH / 16) + wg;   // z column tile (gate-major)

    for (int c = 0; c < NCHUNK; ++c) {
      // pipeline: issue next chunk before waiting on the current one
      if (c + 1 < NCHUNK) {
        const unsigned short* nsrc = (c + 1 < 4) ? (xt + (c + 1) * KCHUNK)
                                                 : (hcur + (c + 1 - 4) * KCHUNK);
        stage_issue(aoff[(c + 1) & 1], nsrc);
        wait_asynccnt<4>();           // in-order done => chunk c landed
      } else if (t + 1 < T_STEPS) {
        // cross-step: next step's x chunk 0 is h-independent, issue it now
        stage_issue(aoff[0], xt + BB * HH);
        wait_asynccnt<4>();
      } else {
        wait_asynccnt<0>();
      }
      __syncthreads();                // all waves' chunk c staged

      const unsigned short (*abuf)[KCHUNK + APAD] = lds_a[c & 1];
#pragma unroll
      for (int kt = 0; kt < KCHUNK / 32; ++kt) {
        const int kti = c * (KCHUNK / 32) + kt;
        __builtin_prefetch((const void*)(wc + ((size_t)((kti + 1) * 256 + ntile) * 512)
                                            + lane * 16), 0, 3);
        bf16x16 afr = load_a_frag(abuf, arow, kt * 32, lane);
        bf16x16 bfr = load_b_frag(wc, (size_t)kti * 256 + ntile, lane);
        acc = __builtin_amdgcn_wmma_f32_16x16x32_bf16(
            false, afr, false, bfr, (short)0, acc, false, false);
      }
      __syncthreads();                // reads done before buffer is rewritten
    }

    // spill z tile: C layout vgpr k, lane L -> M=k+8*(L>>4), N=L&15
    {
      int rbase = rHalf * 16 + ((lane >> 4) << 3);
      int col   = lane & 15;
#pragma unroll
      for (int k = 0; k < 8; ++k) lds_z[gGate][rbase + k][col] = acc[k];
    }
    __syncthreads();

    // registerized gate update for the thread's 2 owned elements
#pragma unroll
    for (int u = 0; u < 2; ++u) {
      float ii = lds_z[0][eb[u]][ejj[u]] + gzi[u];
      float ff = lds_z[1][eb[u]][ejj[u]] + gzf[u];
      float oo = lds_z[2][eb[u]][ejj[u]] + gzo[u];
      float cc = lds_z[3][eb[u]][ejj[u]] + gzc[u];
      float c_ = sigf(ii) * tanhf(cc);
      float hn = sigf(ff) * hreg[u] + av[u] * (sigf(oo) * tanhf(c_));
      hreg[u]  = hn;
      if (t == T_STEPS - 1) creg[u] = c_;
      hnxt[eb[u] * HH + ej[u]] = f2bf(hn);
    }
    grid_barrier(bar);
  }

  // -------------------- Globel_state: zg = h_fin @ Wgh + gzg ---------------
  const unsigned short* hfin = hbf0;  // T_STEPS even -> final h in buffer 0
  f32x8 acc = {};
  const int g3 = wave >> 1;           // waves 0..5 -> gates 0..2
  const int r3 = wave & 1;
  const int arow3 = r3 * 16 + (lane & 15);

  stage_issue(aoff[0], hfin);
  for (int c = 0; c < 4; ++c) {
    if (c + 1 < 4) {
      stage_issue(aoff[(c + 1) & 1], hfin + (c + 1) * KCHUNK);
      wait_asynccnt<4>();
    } else {
      wait_asynccnt<0>();
    }
    __syncthreads();
    if (wave < 6) {
      const unsigned short (*abuf)[KCHUNK + APAD] = lds_a[c & 1];
      const int ntile = g3 * (HH / 16) + wg;
#pragma unroll
      for (int kt = 0; kt < KCHUNK / 32; ++kt) {
        const int kti = c * (KCHUNK / 32) + kt;
        bf16x16 afr = load_a_frag(abuf, arow3, kt * 32, lane);
        bf16x16 bfr = load_b_frag(wgh, (size_t)kti * 192 + ntile, lane);
        acc = __builtin_amdgcn_wmma_f32_16x16x32_bf16(
            false, afr, false, bfr, (short)0, acc, false, false);
      }
    }
    __syncthreads();
  }
  if (wave < 6) {
    int rbase = r3 * 16 + ((lane >> 4) << 3);
    int col   = lane & 15;
#pragma unroll
    for (int k = 0; k < 8; ++k) lds_z[g3][rbase + k][col] = acc[k];
  }
  __syncthreads();

#pragma unroll
  for (int u = 0; u < 2; ++u) {
    int b = eb[u], jj = ejj[u], j = ej[u];
    float ig  = lds_z[0][b][jj] + gzg[j];
    float fg  = lds_z[1][b][jj] + gzg[HH + j];
    float og  = lds_z[2][b][jj] + gzg[2 * HH + j];
    float cgn = sigf(fg) * cg0[j] + sigf(ig) * tanhf(creg[u]);
    out[b * HH + j]           = sigf(og) * tanhf(cgn);
    out[BB * HH + b * HH + j] = cgn;
  }
}

// ---------------------------------------------------------------------------
extern "C" void kernel_launch(void* const* d_in, const int* in_sizes, int n_in,
                              void* d_out, int out_size, void* d_ws, size_t ws_size,
                              hipStream_t stream) {
  (void)in_sizes; (void)n_in; (void)out_size; (void)ws_size;
  const float* x   = (const float*)d_in[0];   // [T,B,IN]
  const float* a   = (const float*)d_in[1];   // [H]
  const float* Wx  = (const float*)d_in[2];   // [IN,4H]
  const float* Wh  = (const float*)d_in[3];   // [H,4H]
  const float* Wg  = (const float*)d_in[4];   // [H,4H]
  const float* b   = (const float*)d_in[5];   // [4H]
  const float* Wgh = (const float*)d_in[6];   // [H,3H]
  const float* Wgg = (const float*)d_in[7];   // [H,3H]
  const float* bg  = (const float*)d_in[8];   // [3H]
  const float* g0  = (const float*)d_in[9];   // [H]
  const float* cg0 = (const float*)d_in[10];  // [H]
  float* out = (float*)d_out;

  // workspace layout (bytes) — total ~90.4 MB
  char* ws = (char*)d_ws;
  const size_t WC_OFF   = 0;                                   // 2048*4096*2
  const size_t WGH_OFF  = WC_OFF   + (size_t)2048 * 4096 * 2;  // 1024*3072*2
  const size_t GZ_OFF   = WGH_OFF  + (size_t)1024 * 3072 * 2;  // 4096*4
  const size_t GZG_OFF  = GZ_OFF   + (size_t)4096 * 4;         // 3072*4
  const size_t XBF_OFF  = GZG_OFF  + (size_t)3072 * 4;         // T*B*IN*2
  const size_t HBF0_OFF = XBF_OFF  + (size_t)T_STEPS * BB * HH * 2;
  const size_t HBF1_OFF = HBF0_OFF + (size_t)BB * HH * 2;
  const size_t BAR_OFF  = HBF1_OFF + (size_t)BB * HH * 2;

  unsigned short* wcP   = (unsigned short*)(ws + WC_OFF);
  unsigned short* wghP  = (unsigned short*)(ws + WGH_OFF);
  float*          gzP   = (float*)(ws + GZ_OFF);
  float*          gzgP  = (float*)(ws + GZG_OFF);
  unsigned short* xbfP  = (unsigned short*)(ws + XBF_OFF);
  unsigned short* hbf0P = (unsigned short*)(ws + HBF0_OFF);
  unsigned short* hbf1P = (unsigned short*)(ws + HBF1_OFF);
  int*            barP  = (int*)(ws + BAR_OFF);

  const int nx = T_STEPS * BB * HH;  // 33,554,432

  init_state_kernel<<<(BB * HH + 255) / 256, 256, 0, stream>>>(hbf0P, hbf1P, barP);
  f32_to_bf16_kernel<<<nx / 256, 256, 0, stream>>>(x, xbfP, nx);
  pack_b_tiles_kernel<<<64 * 256, 256, 0, stream>>>(Wx, Wh, 1024, 4096, wcP);
  pack_b_tiles_kernel<<<32 * 192, 256, 0, stream>>>(Wgh, Wgh, 2048, 3072, wghP);
  colsum_kernel<<<(4096 + 255) / 256, 256, 0, stream>>>(Wg,  b,  g0, gzP,  1024, 4096);
  colsum_kernel<<<(3072 + 255) / 256, 256, 0, stream>>>(Wgg, bg, g0, gzgP, 1024, 3072);
  gr_scan_kernel<<<NWG, 256, 0, stream>>>(xbfP, a, wcP, wghP, gzP, gzgP,
                                          hbf0P, hbf1P, cg0, out, barP);
}
